// RSSM_89378269430173
// MI455X (gfx1250) — matrix operand order repaired and checked by reading, not compiled, not run on hardware
//
#include <hip/hip_runtime.h>
#include <cstddef>

// ---------------- types ----------------
typedef __bf16 bf16_t;
typedef __attribute__((ext_vector_type(16))) __bf16 v16bf;
typedef __attribute__((ext_vector_type(8)))  __bf16 v8bf;
typedef __attribute__((ext_vector_type(8)))  float  v8f;
typedef __attribute__((ext_vector_type(4)))  float  v4f;

// ---------------- problem constants ----------------
#define Bsz 512
#define Tt  64
#define Ss  32
#define Dd  200
#define Hh  200
#define Aa  3
#define Ee  1024
#define OUTW 592           // 2*(3S+D)

// GEMM tiling (M=16 rows per workgroup, 8 waves of 32).
// N-tile counts padded to multiples of 8 so every wave runs a fixed,
// guard-free WMMA schedule (pad tiles hit zero-filled weight fragments).
#define NT200P 16          // real 13
#define NT600P 40          // real 38
#define NT64P   8          // real 4
#define KT_IN   2          // K=35   -> 2 k-steps of 32
#define KT_H    7          // K=200  -> 7 k-steps
#define KT_Q   39          // K=1224 -> 39 k-steps
#define KWPAD 224          // padded row width for K=200 activations
#define KWIN   64          // padded row width for [stoch|action] input

// Fast transcendentals: lower to single v_exp_f32 / v_log_f32 TRANS ops
// (TRANS co-executes with WMMA on CDNA5).
__device__ __forceinline__ float elu_(float x){
  return x > 0.f ? x : (__expf(x) - 1.f);
}
__device__ __forceinline__ float sp_(float x){
  return x > 20.f ? x : __logf(1.f + __expf(x));
}
__device__ __forceinline__ float sig_(float x){
  return 1.f / (1.f + __expf(-x));
}
__device__ __forceinline__ float tanh_(float x){
  float xc = fminf(fmaxf(x, -15.f), 15.f);
  float e = __expf(2.f * xc);
  return (e - 1.f) / (e + 1.f);
}

__device__ __forceinline__ v8f wmma_bf16(v16bf a, v16bf b, v8f c) {
  return __builtin_amdgcn_wmma_f32_16x16x32_bf16(
      false, a, false, b, (short)0, c, false, false);
}

__device__ __forceinline__ v16bf cat16(v8bf lo, v8bf hi) {
  return __builtin_shufflevector(lo, hi, 0,1,2,3,4,5,6,7,8,9,10,11,12,13,14,15);
}
__device__ __forceinline__ v8bf zbf8() {
  v8bf r;
#pragma unroll
  for (int j = 0; j < 8; ++j) r[j] = (bf16_t)0.0f;
  return r;
}
// 8 consecutive fp32 -> v8bf (two aligned b128 loads + pack converts)
__device__ __forceinline__ v8bf cvt8(const float* p) {
  v4f f0 = *(const v4f*)p;
  v4f f1 = *(const v4f*)(p + 4);
  v8bf r;
#pragma unroll
  for (int j = 0; j < 4; ++j) { r[j] = (bf16_t)f0[j]; r[j + 4] = (bf16_t)f1[j]; }
  return r;
}

// B fragment: weights pre-packed so a lane loads 16 contiguous bf16 (32B)
__device__ __forceinline__ v16bf bfrag(const bf16_t* __restrict__ W, int KT,
                                       int nt, int kt, int lane) {
  return *(const v16bf*)(W + (((size_t)(nt * KT + kt) * 32 + lane) << 4));
}

// A fragment from zero-padded row-major bf16 LDS (ISA 16-bit A 16x32 layout):
// lane's j<8 half = K[ktbase+kg*8 .. +7], j>=8 half = K[ktbase+16+kg*8 .. +7]
// -> exactly two aligned ds_load_b128 per fragment.
__device__ __forceinline__ v16bf afrag(const bf16_t* lds, int m, int kw,
                                       int ktbase, int kg) {
  const bf16_t* base = lds + m * kw + ktbase + kg * 8;
  v8bf lo = *(const v8bf*)(base);
  v8bf hi = *(const v8bf*)(base + 16);
  return cat16(lo, hi);
}

// ---------------- weight repack: fp32 [N,K] -> bf16 B-fragment layout ----------------
// Zero-fills n >= N and k >= K, so padded tiles are exact zero matrices.
__global__ void convert_w(const float* __restrict__ src, bf16_t* __restrict__ dst,
                          int N, int K, int NT, int KT) {
  int total = NT * KT * 512;
  for (int idx = blockIdx.x * blockDim.x + threadIdx.x; idx < total;
       idx += gridDim.x * blockDim.x) {
    int j    = idx & 15;
    int lane = (idx >> 4) & 31;
    int frag = idx >> 9;
    int kt   = frag % KT;
    int nt   = frag / KT;
    int n    = nt * 16 + (lane & 15);
    int kg   = lane >> 4;
    int k    = kt * 32 + (j < 8 ? kg * 8 + j : 16 + kg * 8 + (j - 8));
    float v  = (n < N && k < K) ? src[(size_t)n * K + k] : 0.f;
    dst[idx] = (bf16_t)v;
  }
}

// ---------------- persistent RSSM kernel: 32 WGs x 256 threads ----------------
// launch_bounds(256, 4): min 4 waves/EU caps VGPRs so weights are NOT
// hoisted/spilled; they stream from L2 every iteration.
__global__ __launch_bounds__(256, 4) void rssm_persistent(
    const float* __restrict__ embeds, const float* __restrict__ actions,
    const float* __restrict__ n_pr, const float* __restrict__ n_po,
    const float* __restrict__ b_in, const float* __restrict__ b_ih,
    const float* __restrict__ b_hh, const float* __restrict__ bp1,
    const float* __restrict__ bp2, const float* __restrict__ bq1,
    const float* __restrict__ bq2,
    const bf16_t* __restrict__ Wg_in, const bf16_t* __restrict__ Wg_ih,
    const bf16_t* __restrict__ Wg_hh, const bf16_t* __restrict__ Wg_p1,
    const bf16_t* __restrict__ Wg_p2, const bf16_t* __restrict__ Wg_q1,
    const bf16_t* __restrict__ Wg_q2, float* __restrict__ out) {

  // ---- LDS pool, 54784B. Overlays:
  //   [0,19200)      sSum  bf16[16][600]  (phases 2-3; r/z cols: gi+gh, n cols: gi=inn)
  //   [19200,25600)  sHn   bf16[16][200]  (phases 2-3; gh for n-gate)
  //   [0,4096)       sPS   f32[16][64]    (phases 5-8; overlay, sSum dead)
  //   [4096,8192)    sQS   f32[16][64]    (phases 7-8; overlay)
  __shared__ __align__(16) char pool[54784];
  bf16_t* sSum     = (bf16_t*)(pool + 0);
  bf16_t* sHn      = (bf16_t*)(pool + 19200);
  float*  sPS      = (float*)(pool + 0);
  float*  sQS      = (float*)(pool + 4096);
  bf16_t* sT1      = (bf16_t*)(pool + 25600);  // [16][224] X / P1 / Q1 (zero-padded)
  bf16_t* sDeterBf = (bf16_t*)(pool + 32768);  // [16][224] bf16 deter (zero-padded)
  float*  sDeter   = (float*)(pool + 39936);   // [16][200] f32 deter (state)
  bf16_t* sXin     = (bf16_t*)(pool + 52736);  // [16][64]  [stoch(32)|act(3)|0] (state)

  const int tid   = threadIdx.x;
  const int lane  = tid & 31;
  const int w     = __builtin_amdgcn_readfirstlane(tid >> 5);  // wave id, SGPR
  const int ml    = lane & 15;         // A-matrix row for this lane
  const int kg    = lane >> 4;         // A-matrix K-half group
  const int nlane = lane & 15;         // C-matrix column within tile
  const int mbase = (lane >> 4) * 8;   // C-matrix row base
  const int m0    = blockIdx.x * 16;   // batch rows [m0, m0+16)

  v8f vzero = {0.f, 0.f, 0.f, 0.f, 0.f, 0.f, 0.f, 0.f};

  // ---- init state + zero pads (pads are never written again) ----
  for (int i = tid; i < 16 * KWIN; i += 256) sXin[i] = (bf16_t)0.0f;
  for (int i = tid; i < 16 * KWPAD; i += 256) {
    sT1[i] = (bf16_t)0.0f;
    sDeterBf[i] = (bf16_t)0.0f;
  }
  for (int i = tid; i < 16 * Dd; i += 256) sDeter[i] = 0.f;
  __syncthreads();

  for (int t = 0; t < Tt; ++t) {
    // Opaque memory clobber: stops LICM from hoisting the (t-invariant)
    // weight-fragment loads out of the time loop (which caused VGPR-MSB
    // allocation + scratch spilling). Weights re-stream from L2 each step.
    asm volatile("" ::: "memory");

    // ============ phase 0: stage prev action into sXin, prefetch next embeds ===
    if (tid < 48) {
      int mm = tid / 3, c = tid % 3;
      float av = 0.f;
      if (t > 0) av = actions[((size_t)(m0 + mm) * (Tt - 1) + (t - 1)) * Aa + c];
      sXin[mm * KWIN + Ss + c] = (bf16_t)av;
    }
    if (t + 1 < Tt) {
      const float* ep = embeds + (((size_t)(m0 + (tid & 15))) * Tt + (t + 1)) * Ee;
      __builtin_prefetch(ep + (tid >> 4) * 64, 0, 1);  // global_prefetch_b8
    }
    __syncthreads();

    // ============ phase 1: X = elu([stoch,a] @ W_in^T + b_in) -> sT1 ============
    {
      v8f acc[2] = {vzero, vzero};
#pragma unroll
      for (int kt = 0; kt < KT_IN; ++kt) {
        v16bf a = afrag(sXin, ml, KWIN, kt * 32, kg);
#pragma unroll
        for (int i = 0; i < 2; ++i)
          acc[i] = wmma_bf16(a, bfrag(Wg_in, KT_IN, w + i * 8, kt, lane), acc[i]);
      }
#pragma unroll
      for (int i = 0; i < 2; ++i) {
        int n = (w + i * 8) * 16 + nlane;
        if (n < Hh) {
          float bb = b_in[n];
#pragma unroll
          for (int r = 0; r < 8; ++r)
            sT1[(mbase + r) * KWPAD + n] = (bf16_t)elu_(acc[i][r] + bb);
        }
      }
    }
    __syncthreads();

    // ====== phase 2: GI = X@W_ih^T+b_ih, GH = deter@W_hh^T+b_hh -> sSum/sHn =====
    {
      v8f acci[5], acch[5];
#pragma unroll
      for (int i = 0; i < 5; ++i) { acci[i] = vzero; acch[i] = vzero; }
#pragma unroll
      for (int kt = 0; kt < KT_H; ++kt) {
        v16bf ax = afrag(sT1,      ml, KWPAD, kt * 32, kg);
        v16bf ad = afrag(sDeterBf, ml, KWPAD, kt * 32, kg);
#pragma unroll
        for (int i = 0; i < 5; ++i) {
          int nt = w + i * 8;   // 0..39, all valid in padded weight arrays
          acci[i] = wmma_bf16(ax, bfrag(Wg_ih, KT_H, nt, kt, lane), acci[i]);
          acch[i] = wmma_bf16(ad, bfrag(Wg_hh, KT_H, nt, kt, lane), acch[i]);
        }
      }
#pragma unroll
      for (int i = 0; i < 5; ++i) {
        int nt = w + i * 8;
        int n  = nt * 16 + nlane;              // tile entirely on one side of 400
        if (nt < 25) {                          // r/z columns: store gi+gh (n<400)
          float bi = b_ih[n], bh = b_hh[n];
#pragma unroll
          for (int r = 0; r < 8; ++r)
            sSum[(mbase + r) * 600 + n] =
                (bf16_t)((acci[i][r] + bi) + (acch[i][r] + bh));
        } else if (n < 600) {                   // n-gate columns: inn and hn
          float bi = b_ih[n], bh = b_hh[n];
#pragma unroll
          for (int r = 0; r < 8; ++r) {
            sSum[(mbase + r) * 600 + n]        = (bf16_t)(acci[i][r] + bi);
            sHn[(mbase + r) * 200 + (n - 400)] = (bf16_t)(acch[i][r] + bh);
          }
        }
      }
    }
    __syncthreads();

    // ============ phase 3: GRU gates -> deter_new (fp32 + bf16 mirror) ==========
    for (int idx = tid; idx < 16 * Dd; idx += 256) {
      int mm = idx / Dd, c = idx % Dd;
      float r   = sig_((float)sSum[mm * 600 + c]);
      float z   = sig_((float)sSum[mm * 600 + c + 200]);
      float inn = (float)sSum[mm * 600 + c + 400];
      float hn  = (float)sHn[mm * 200 + c];
      float nn  = tanh_(inn + r * hn);
      float dnew = (1.f - z) * nn + z * sDeter[mm * Dd + c];
      sDeter[mm * Dd + c]      = dnew;
      sDeterBf[mm * KWPAD + c] = (bf16_t)dnew;
    }
    __syncthreads();

    // ============ phase 4: P1 = elu(deter@Wp1^T + bp1) -> sT1 ===================
    {
      v8f acc[2] = {vzero, vzero};
#pragma unroll
      for (int kt = 0; kt < KT_H; ++kt) {
        v16bf ad = afrag(sDeterBf, ml, KWPAD, kt * 32, kg);
#pragma unroll
        for (int i = 0; i < 2; ++i)
          acc[i] = wmma_bf16(ad, bfrag(Wg_p1, KT_H, w + i * 8, kt, lane), acc[i]);
      }
#pragma unroll
      for (int i = 0; i < 2; ++i) {
        int n = (w + i * 8) * 16 + nlane;
        if (n < Hh) {
          float bb = bp1[n];
#pragma unroll
          for (int r = 0; r < 8; ++r)
            sT1[(mbase + r) * KWPAD + n] = (bf16_t)elu_(acc[i][r] + bb);
        }
      }
    }
    __syncthreads();

    // ============ phase 5: PS = P1 @ Wp2^T + bp2 -> sPS (all waves, store w<4) ==
    {
      v8f acc = vzero;
#pragma unroll
      for (int kt = 0; kt < KT_H; ++kt) {
        v16bf a = afrag(sT1, ml, KWPAD, kt * 32, kg);
        acc = wmma_bf16(a, bfrag(Wg_p2, KT_H, w, kt, lane), acc);
      }
      if (w < 4) {
        int n = w * 16 + nlane;
        float bb = bp2[n];
#pragma unroll
        for (int r = 0; r < 8; ++r) sPS[(mbase + r) * 64 + n] = acc[r] + bb;
      }
    }
    __syncthreads();

    // ============ phase 6: Q1 = elu([deter,e]@Wq1^T + bq1) -> sT1 ===============
    {
      const float* erow = embeds + (((size_t)(m0 + ml)) * Tt + t) * Ee;
      v8f acc[2] = {vzero, vzero};
      // kt 0..5: pure deter (branch-free LDS vector loads)
#pragma unroll
      for (int kt = 0; kt < 6; ++kt) {
        v16bf a = afrag(sDeterBf, ml, KWPAD, kt * 32, kg);
#pragma unroll
        for (int i = 0; i < 2; ++i)
          acc[i] = wmma_bf16(a, bfrag(Wg_q1, KT_Q, w + i * 8, kt, lane), acc[i]);
      }
      // kt == 6: straddles deter/embed boundary (K 192..223)
      {
        v8bf lo = (kg == 0) ? *(const v8bf*)(sDeterBf + ml * KWPAD + 192)
                            : cvt8(erow + 0);
        v8bf hi = cvt8(erow + 8 + kg * 8);
        v16bf a = cat16(lo, hi);
#pragma unroll
        for (int i = 0; i < 2; ++i)
          acc[i] = wmma_bf16(a, bfrag(Wg_q1, KT_Q, w + i * 8, 6, lane), acc[i]);
      }
      // kt 7..37: pure embeds (two aligned float4-pair loads + pack cvt)
      for (int kt = 7; kt < 38; ++kt) {
        const float* p = erow + (kt * 32 - Dd) + kg * 8;
        v16bf a = cat16(cvt8(p), cvt8(p + 16));
#pragma unroll
        for (int i = 0; i < 2; ++i)
          acc[i] = wmma_bf16(a, bfrag(Wg_q1, KT_Q, w + i * 8, kt, lane), acc[i]);
      }
      // kt == 38: tail (K 1216..1247, valid only K<1224)
      {
        v8bf lo = (kg == 0) ? cvt8(erow + 1016) : zbf8();
        v16bf a = cat16(lo, zbf8());
#pragma unroll
        for (int i = 0; i < 2; ++i)
          acc[i] = wmma_bf16(a, bfrag(Wg_q1, KT_Q, w + i * 8, 38, lane), acc[i]);
      }
#pragma unroll
      for (int i = 0; i < 2; ++i) {
        int n = (w + i * 8) * 16 + nlane;
        if (n < Hh) {
          float bb = bq1[n];
#pragma unroll
          for (int r = 0; r < 8; ++r)
            sT1[(mbase + r) * KWPAD + n] = (bf16_t)elu_(acc[i][r] + bb);
        }
      }
    }
    __syncthreads();

    // ============ phase 7: QS = Q1 @ Wq2^T + bq2 -> sQS (all waves, store w<4) ==
    {
      v8f acc = vzero;
#pragma unroll
      for (int kt = 0; kt < KT_H; ++kt) {
        v16bf a = afrag(sT1, ml, KWPAD, kt * 32, kg);
        acc = wmma_bf16(a, bfrag(Wg_q2, KT_H, w, kt, lane), acc);
      }
      if (w < 4) {
        int n = w * 16 + nlane;
        float bb = bq2[n];
#pragma unroll
        for (int r = 0; r < 8; ++r) sQS[(mbase + r) * 64 + n] = acc[r] + bb;
      }
    }
    __syncthreads();

    // ============ phase 8: reparam + output write + stoch update ================
    for (int idx = tid; idx < 16 * OUTW; idx += 256) {
      int mm = idx / OUTW, j = idx % OUTW;
      int b = m0 + mm;
      size_t obase = ((size_t)b * Tt + t) * OUTW;
      float val;
      if (j < 32) {                            // q_mean
        val = sQS[mm * 64 + j];
      } else if (j < 64) {                     // q_std
        val = sp_(sQS[mm * 64 + j]) + 0.1f;
      } else if (j < 96) {                     // q_stoch (+ state update)
        int s = j - 64;
        float mean = sQS[mm * 64 + s];
        float std  = sp_(sQS[mm * 64 + 32 + s]) + 0.1f;
        float qs   = mean + std * n_po[((size_t)b * Tt + t) * Ss + s];
        sXin[mm * KWIN + s] = (bf16_t)qs;      // stoch for next step
        val = qs;
      } else if (j < 296) {                    // deter_new
        val = sDeter[mm * Dd + (j - 96)];
      } else if (j < 328) {                    // p_mean
        val = sPS[mm * 64 + (j - 296)];
      } else if (j < 360) {                    // p_std
        val = sp_(sPS[mm * 64 + 32 + (j - 328)]) + 0.1f;
      } else if (j < 392) {                    // p_stoch
        int s = j - 360;
        float mean = sPS[mm * 64 + s];
        float std  = sp_(sPS[mm * 64 + 32 + s]) + 0.1f;
        val = mean + std * n_pr[((size_t)b * Tt + t) * Ss + s];
      } else {                                 // deter_new (again)
        val = sDeter[mm * Dd + (j - 392)];
      }
      out[obase + j] = val;
    }
    __syncthreads();
  }
}

// ---------------- host launch ----------------
extern "C" void kernel_launch(void* const* d_in, const int* in_sizes, int n_in,
                              void* d_out, int out_size, void* d_ws, size_t ws_size,
                              hipStream_t stream) {
  const float* embeds  = (const float*)d_in[0];
  const float* actions = (const float*)d_in[1];
  const float* n_pr    = (const float*)d_in[2];
  const float* n_po    = (const float*)d_in[3];
  const float* W_in = (const float*)d_in[4];  const float* b_in = (const float*)d_in[5];
  const float* W_ih = (const float*)d_in[6];  const float* b_ih = (const float*)d_in[7];
  const float* W_hh = (const float*)d_in[8];  const float* b_hh = (const float*)d_in[9];
  const float* Wp1  = (const float*)d_in[10]; const float* bp1  = (const float*)d_in[11];
  const float* Wp2  = (const float*)d_in[12]; const float* bp2  = (const float*)d_in[13];
  const float* Wq1  = (const float*)d_in[14]; const float* bq1  = (const float*)d_in[15];
  const float* Wq2  = (const float*)d_in[16]; const float* bq2  = (const float*)d_in[17];
  float* out = (float*)d_out;

  char* ws = (char*)d_ws;
  size_t off = 0;
  auto carve = [&](size_t nfrag) -> bf16_t* {
    bf16_t* p = (bf16_t*)(ws + off);
    off += ((nfrag * 512 * sizeof(bf16_t)) + 255) & ~(size_t)255;
    return p;
  };
  bf16_t* Wg_in = carve((size_t)NT200P * KT_IN);
  bf16_t* Wg_ih = carve((size_t)NT600P * KT_H);
  bf16_t* Wg_hh = carve((size_t)NT600P * KT_H);
  bf16_t* Wg_p1 = carve((size_t)NT200P * KT_H);
  bf16_t* Wg_p2 = carve((size_t)NT64P  * KT_H);
  bf16_t* Wg_q1 = carve((size_t)NT200P * KT_Q);
  bf16_t* Wg_q2 = carve((size_t)NT64P  * KT_H);

  auto cvt = [&](const float* src, bf16_t* dst, int N, int K, int NT, int KT) {
    int total = NT * KT * 512;
    convert_w<<<(total + 255) / 256, 256, 0, stream>>>(src, dst, N, K, NT, KT);
  };
  cvt(W_in, Wg_in, Hh,     Ss + Aa, NT200P, KT_IN);
  cvt(W_ih, Wg_ih, 3 * Dd, Hh,      NT600P, KT_H);
  cvt(W_hh, Wg_hh, 3 * Dd, Dd,      NT600P, KT_H);
  cvt(Wp1,  Wg_p1, Hh,     Dd,      NT200P, KT_H);
  cvt(Wp2,  Wg_p2, 2 * Ss, Hh,      NT64P,  KT_H);
  cvt(Wq1,  Wg_q1, Hh,     Dd + Ee, NT200P, KT_Q);
  cvt(Wq2,  Wg_q2, 2 * Ss, Hh,      NT64P,  KT_H);

  rssm_persistent<<<Bsz / 16, 256, 0, stream>>>(
      embeds, actions, n_pr, n_po,
      b_in, b_ih, b_hh, bp1, bp2, bq1, bq2,
      Wg_in, Wg_ih, Wg_hh, Wg_p1, Wg_p2, Wg_q1, Wg_q2, out);
}
